// Matformer4Multi_50508815401617
// MI455X (gfx1250) — compile-verified, matrix operand order checked by reading, not run on hardware
//
#include <hip/hip_runtime.h>
#include <hip/hip_bf16.h>
#include <stdint.h>

// ---------------------------------------------------------------- constants
#define NN      16000
#define NE      80000
#define NG      256
#define FDIM    128
#define HH      4
#define CC      128
#define HC      512          // H*C
#define TC      384          // 3*C
#define NLAYERS 5
#define ATOMF   92

#define USE_ASYNC_COPY 1

static __device__ __constant__ float kScale = 0.05103103630798287f; // 1/sqrt(3*128)

typedef unsigned short u16;
typedef __attribute__((ext_vector_type(16))) __bf16        v16bf;
typedef __attribute__((ext_vector_type(8)))  float         v8f;
typedef __attribute__((ext_vector_type(8)))  unsigned int  v8u;
typedef __attribute__((address_space(3)))    u16           lds_u16;

// ---------------------------------------------------------------- helpers
__device__ __forceinline__ u16 f2bf(float f) {
  unsigned int u = __float_as_uint(f);
  u += 0x7FFFu + ((u >> 16) & 1u);            // round-to-nearest-even
  return (u16)(u >> 16);
}
__device__ __forceinline__ float bf2f(u16 h) {
  return __uint_as_float(((unsigned int)h) << 16);
}
// pack two f32 -> packed bf16x2 (v_cvt_pk_bf16_f32 when available)
__device__ __forceinline__ unsigned packbf(float lo, float hi) {
#if __has_builtin(__builtin_amdgcn_cvt_pk_bf16_f32)
  typedef __attribute__((ext_vector_type(2))) __bf16 v2bf;
  v2bf r = __builtin_amdgcn_cvt_pk_bf16_f32(lo, hi);
  return __builtin_bit_cast(unsigned, r);
#else
  return (unsigned)f2bf(lo) | ((unsigned)f2bf(hi) << 16);
#endif
}
__device__ __forceinline__ float sigmoidf_(float x) { return 1.f / (1.f + __expf(-x)); }

// 16-byte global->LDS copy; async (ASYNCcnt) on CDNA5 when enabled.
__device__ __forceinline__ void cp16_g2l(u16* ldsp, const u16* gp) {
#if USE_ASYNC_COPY
  unsigned loff = (unsigned)(size_t)(lds_u16*)ldsp;
  unsigned long long ga = (unsigned long long)(uintptr_t)gp;
  asm volatile("global_load_async_to_lds_b128 %0, %1, off"
               :: "v"(loff), "v"(ga)
               : "memory");
#else
  *(uint4*)ldsp = *(const uint4*)gp;
#endif
}
__device__ __forceinline__ void wait_async() {
#if USE_ASYNC_COPY
  asm volatile("s_wait_asynccnt 0" ::: "memory");
#endif
}

// Load one 16x32 bf16 fragment from LDS (K-contiguous rows).
// Lane layout per CDNA5 ISA 7.12.2: lanes<16 hold K {kb+0..7, kb+16..23},
// lanes>=16 hold K {kb+8..15, kb+24..31}; two 16B contiguous runs -> b128 loads.
__device__ __forceinline__ v16bf ld_frag(const u16* rowp, int kb, int half) {
  const u16* p = rowp + kb + half * 8;
  uint4 lo = *(const uint4*)(p);
  uint4 hi = *(const uint4*)(p + 16);
  v8u t;
  t[0] = lo.x; t[1] = lo.y; t[2] = lo.z; t[3] = lo.w;
  t[4] = hi.x; t[5] = hi.y; t[6] = hi.z; t[7] = hi.w;
  return __builtin_bit_cast(v16bf, t);
}

__device__ __forceinline__ v8f wmma_bf16(v16bf a, v16bf b, v8f c) {
  return __builtin_amdgcn_wmma_f32_16x16x32_bf16(false, a, false, b, (short)0, c,
                                                 false, false);
}

// ---------------------------------------------------------------- weight prep
// Wt[n][k] = bf16(W[k][n]); rows padded to Kpad (zero fill).
__global__ void mf_tr(const float* __restrict__ W, u16* __restrict__ Wt,
                      int K, int N, int Kpad) {
  int i = blockIdx.x * blockDim.x + threadIdx.x;
  if (i >= N * Kpad) return;
  int n = i / Kpad, k = i % Kpad;
  float v = (k < K) ? W[(size_t)k * N + n] : 0.f;
  Wt[(size_t)n * Kpad + k] = f2bf(v);
}

// ---------------------------------------------------------------- generic GEMM
// C[M,N] = act(A[M,K] @ B[K,N] + bias), B given pre-transposed bf16 Bt[n][k]
// with row stride bstride. BM=32, BN=128, 256 threads (8 waves).
// ACT: 0 none, 1 softplus, 2 silu. OBF16: write bf16 output.
template <int ACT, bool OBF16>
__global__ __launch_bounds__(256) void mf_gemm(const float* __restrict__ A,
                                               const u16* __restrict__ Bt,
                                               const float* __restrict__ bias,
                                               void* __restrict__ Cout,
                                               int M, int N, int K, int bstride) {
  __shared__ __align__(16) u16 Asm[32 * 40];
  __shared__ __align__(16) u16 Bsm[128 * 40];   // [n][k]

  const int tid  = threadIdx.x;
  const int w    = tid >> 5;
  const int lane = tid & 31;
  const int l16  = lane & 15;
  const int half = lane >> 4;
  const int m0   = blockIdx.y * 32;
  const int n0   = blockIdx.x * 128;

  v8f acc0 = {}; v8f acc1 = {};
  const int ksteps = (K + 31) / 32;

  for (int ks = 0; ks < ksteps; ++ks) {
    const int kbase = ks * 32;
    // stage A 32x32: f32 -> packed bf16 pairs
#pragma unroll
    for (int r2 = 0; r2 < 2; ++r2) {
      int flat = r2 * 256 + tid;                 // 0..511 pairs
      int rr = flat >> 4, kk = (flat & 15) * 2;
      int gr = m0 + rr, gk = kbase + kk;
      float v0 = (gr < M && gk < K)     ? A[(size_t)gr * K + gk]     : 0.f;
      float v1 = (gr < M && gk + 1 < K) ? A[(size_t)gr * K + gk + 1] : 0.f;
      *(unsigned*)&Asm[rr * 40 + kk] = packbf(v0, v1);
    }
    // stage B 128x32 bf16: pure 16B copies (async global->LDS)
#pragma unroll
    for (int r2 = 0; r2 < 2; ++r2) {
      int flat = r2 * 256 + tid;                 // 0..511 chunks of 8 bf16
      int n = flat >> 2, ch = (flat & 3) * 8;
      cp16_g2l(&Bsm[n * 40 + ch], &Bt[(size_t)(n0 + n) * bstride + kbase + ch]);
    }
    if (kbase + 32 < K)
      __builtin_prefetch(&Bt[(size_t)(n0 + (tid >> 2)) * bstride + kbase + 32], 0, 0);
    wait_async();
    __syncthreads();

    v16bf bf = ld_frag(&Bsm[(w * 16 + l16) * 40], 0, half);
    v16bf a0 = ld_frag(&Asm[l16 * 40], 0, half);
    v16bf a1 = ld_frag(&Asm[(16 + l16) * 40], 0, half);
    acc0 = wmma_bf16(a0, bf, acc0);
    acc1 = wmma_bf16(a1, bf, acc1);
    __syncthreads();
  }

  const int c = n0 + w * 16 + l16;
  const float bv = bias ? bias[c] : 0.f;
#pragma unroll
  for (int t2 = 0; t2 < 2; ++t2) {
    v8f acc = t2 ? acc1 : acc0;
#pragma unroll
    for (int j = 0; j < 8; ++j) {
      int r = m0 + t2 * 16 + j + half * 8;
      if (r < M) {
        float v = acc[j] + bv;
        if (ACT == 1) v = (v > 20.f) ? v : log1pf(__expf(v));
        if (ACT == 2) v = v * sigmoidf_(v);
        if (OBF16) ((u16*)Cout)[(size_t)r * N + c] = f2bf(v);
        else       ((float*)Cout)[(size_t)r * N + c] = v;
      }
    }
  }
}

// ---------------------------------------------------------------- fused edge kernel
// Per block: 8 edges -> 32 (edge,head) rows, row r = localEdge*4 + head.
//  1) msg = [v_dst|v_src|ea] (raw bf16 u32 copies) + alpha row stats over 384
//  2) GEMM1: msg @ Wmu (+bmu), gated by sigmoid(LN(alpha))  -> M2 (bf16 LDS)
//  3) GEMM2: M2 @ Wml (+bml)                                 -> o2 (f32 LDS)
//  4) LN over 128, scale/shift, atomic scatter into agg[dst]
__global__ __launch_bounds__(256) void mf_edge(
    const int* __restrict__ src, const int* __restrict__ dst,
    const u16* __restrict__ qb, const u16* __restrict__ kmat,
    const u16* __restrict__ vb, const u16* __restrict__ eab,
    const u16* __restrict__ WmuT, const float* __restrict__ bmu,
    const u16* __restrict__ WmlT, const float* __restrict__ bml,
    const float* __restrict__ lag, const float* __restrict__ lab,
    const float* __restrict__ lmg, const float* __restrict__ lmb,
    float* __restrict__ agg) {
  __shared__ __align__(16) u16 Amsg[32 * 400];  // msg tile, later aliased as f32 o2
  __shared__ __align__(16) u16 M2s[32 * 400];   // gated GEMM1 output
  __shared__ __align__(16) u16 BsT[128 * 40];   // weight tile [n][k]
  __shared__ float s1[32], s2[32], rm[32], rs[32];
  __shared__ int edstS[8], esrcS[8];

  const int tid  = threadIdx.x;
  const int w    = tid >> 5;
  const int lane = tid & 31;
  const int l16  = lane & 15;
  const int half = lane >> 4;
  const int e0   = blockIdx.x * 8;

  if (tid < 8) { edstS[tid] = dst[e0 + tid]; esrcS[tid] = src[e0 + tid]; }
  if (tid < 32) { s1[tid] = 0.f; s2[tid] = 0.f; }
  __syncthreads();

  // ---- stage A: msg tile (u32 pair copies) + alpha statistics --------------
  {
    const int r  = tid >> 3;      // 0..31
    const int ji = tid & 7;
    const int eL = r >> 2, hg = r & 3;
    const int d = edstS[eL], s = esrcS[eL];
    const u16* vD  = vb   + (size_t)d * HC + hg * CC;
    const u16* vS  = vb   + (size_t)s * HC + hg * CC;
    const u16* kD  = kmat + (size_t)d * HC + hg * CC;
    const u16* kS  = kmat + (size_t)s * HC + hg * CC;
    const u16* eaR = eab  + (size_t)(e0 + eL) * HC + hg * CC;
    const u16* qD  = qb   + (size_t)d * HC + hg * CC;
    float p1 = 0.f, p2 = 0.f;
    for (int pp = ji; pp < 192; pp += 8) {       // bf16 pairs
      int jj = pp * 2;
      int seg = jj >> 7, c = jj & 127;
      unsigned mpair, kpair;
      if (seg == 0)      { mpair = *(const unsigned*)(vD + c); kpair = *(const unsigned*)(kD + c); }
      else if (seg == 1) { mpair = *(const unsigned*)(vS + c); kpair = *(const unsigned*)(kS + c); }
      else               { mpair = *(const unsigned*)(eaR + c); kpair = mpair; }
      *(unsigned*)&Amsg[r * 400 + jj] = mpair;   // msg tile: raw bf16 copy
      unsigned qpair = *(const unsigned*)(qD + c);
      float a0 = bf2f((u16)qpair) * bf2f((u16)kpair) * kScale;
      float a1 = bf2f((u16)(qpair >> 16)) * bf2f((u16)(kpair >> 16)) * kScale;
      p1 += a0 + a1;
      p2 += a0 * a0 + a1 * a1;
    }
    atomicAdd(&s1[r], p1);
    atomicAdd(&s2[r], p2);
  }
  __syncthreads();
  if (tid < 32) {
    float mean = s1[tid] * (1.f / TC);
    float var  = s2[tid] * (1.f / TC) - mean * mean;
    rm[tid] = mean;
    rs[tid] = rsqrtf(var + 1e-5f);
  }
  __syncthreads();

  // ---- GEMM1: [32,384] @ Wmu[384,384], gated, 3 N-chunks of 128 ------------
  for (int nc = 0; nc < 3; ++nc) {
    const int ncb = nc * 128;
    v8f acc0 = {}; v8f acc1 = {};
    for (int ks = 0; ks < 12; ++ks) {
      const int kbase = ks * 32;
#pragma unroll
      for (int r2 = 0; r2 < 2; ++r2) {
        int flat = r2 * 256 + tid;
        int n = flat >> 2, ch = (flat & 3) * 8;
        cp16_g2l(&BsT[n * 40 + ch], &WmuT[(size_t)(ncb + n) * TC + kbase + ch]);
      }
      wait_async();
      __syncthreads();
      v16bf bf = ld_frag(&BsT[(w * 16 + l16) * 40], 0, half);
      v16bf a0 = ld_frag(&Amsg[l16 * 400], kbase, half);
      v16bf a1 = ld_frag(&Amsg[(16 + l16) * 400], kbase, half);
      acc0 = wmma_bf16(a0, bf, acc0);
      acc1 = wmma_bf16(a1, bf, acc1);
      __syncthreads();
    }
    // epilogue: bias + sigmoid(LN(alpha)) gate, write bf16 M2
    const int n = ncb + w * 16 + l16;
    const float bmuv = bmu[n], gg = lag[n], gb = lab[n];
    const int seg = n >> 7, c = n & 127;
#pragma unroll
    for (int t2 = 0; t2 < 2; ++t2) {
      v8f acc = t2 ? acc1 : acc0;
#pragma unroll
      for (int j = 0; j < 8; ++j) {
        int r = t2 * 16 + j + half * 8;
        int eL = r >> 2, hg = r & 3;
        int d = edstS[eL], s = esrcS[eL];
        size_t off = (size_t)hg * CC + c;
        float qv = bf2f(qb[(size_t)d * HC + off]);
        float kv = (seg == 0) ? bf2f(kmat[(size_t)d * HC + off])
                 : (seg == 1) ? bf2f(kmat[(size_t)s * HC + off])
                              : bf2f(eab[(size_t)(e0 + eL) * HC + off]);
        float av   = qv * kv * kScale;
        float lnv  = (av - rm[r]) * rs[r] * gg + gb;
        float gate = sigmoidf_(lnv);
        M2s[r * 400 + n] = f2bf((acc[j] + bmuv) * gate);
      }
    }
    __syncthreads();
  }

  // ---- GEMM2: [32,384] @ Wml[384,128] --------------------------------------
  float* o2 = reinterpret_cast<float*>(Amsg);   // Amsg dead; reuse as f32 [32][132]
  {
    v8f c0 = {}; v8f c1 = {};
    for (int ks = 0; ks < 12; ++ks) {
      const int kbase = ks * 32;
#pragma unroll
      for (int r2 = 0; r2 < 2; ++r2) {
        int flat = r2 * 256 + tid;
        int n = flat >> 2, ch = (flat & 3) * 8;
        cp16_g2l(&BsT[n * 40 + ch], &WmlT[(size_t)n * TC + kbase + ch]);
      }
      wait_async();
      __syncthreads();
      v16bf bf = ld_frag(&BsT[(w * 16 + l16) * 40], 0, half);
      v16bf a0 = ld_frag(&M2s[l16 * 400], kbase, half);
      v16bf a1 = ld_frag(&M2s[(16 + l16) * 400], kbase, half);
      c0 = wmma_bf16(a0, bf, c0);
      c1 = wmma_bf16(a1, bf, c1);
      __syncthreads();
    }
    const int n = w * 16 + l16;
    const float bmlv = bml[n];
#pragma unroll
    for (int t2 = 0; t2 < 2; ++t2) {
      v8f acc = t2 ? c1 : c0;
#pragma unroll
      for (int j = 0; j < 8; ++j) {
        int r = t2 * 16 + j + half * 8;
        o2[r * 132 + n] = acc[j] + bmlv;
      }
    }
  }
  __syncthreads();

  // ---- LN over 128 + scatter -----------------------------------------------
  if (tid < 32) { s1[tid] = 0.f; s2[tid] = 0.f; }
  __syncthreads();
  {
    const int r = tid >> 3, ji = tid & 7;
    float p1 = 0.f, p2 = 0.f;
    for (int c = ji; c < CC; c += 8) {
      float v = o2[r * 132 + c];
      p1 += v; p2 += v * v;
    }
    atomicAdd(&s1[r], p1);
    atomicAdd(&s2[r], p2);
  }
  __syncthreads();
  if (tid < 32) {
    float mean = s1[tid] * (1.f / CC);
    float var  = s2[tid] * (1.f / CC) - mean * mean;
    rm[tid] = mean;
    rs[tid] = rsqrtf(var + 1e-5f);
  }
  __syncthreads();
  {
    const int r = tid >> 3, ji = tid & 7;
    const int eL = r >> 2, hg = r & 3;
    const int d = edstS[eL];
    float* dstp = agg + (size_t)d * HC + hg * CC;
    for (int c = ji; c < CC; c += 8) {
      float v = (o2[r * 132 + c] - rm[r]) * rs[r] * lmg[c] + lmb[c];
      atomicAdd(&dstp[c], v);
    }
  }
}

// ---------------------------------------------------------------- small utility kernels
__global__ void mf_zero(float* __restrict__ p, int n) {
  int i = blockIdx.x * blockDim.x + threadIdx.x;
  int stride = gridDim.x * blockDim.x;
  for (; i < n; i += stride) p[i] = 0.f;
}

__global__ void mf_task_add(float* __restrict__ h, const float* __restrict__ temb,
                            const int* __restrict__ task_id) {
  int i = blockIdx.x * blockDim.x + threadIdx.x;
  if (i < NN * FDIM) h[i] += temb[(size_t)task_id[0] * FDIM + (i & 127)];
}

// BatchNorm stats: per-block partial sums over 128 rows x 128 cols.
__global__ __launch_bounds__(256) void mf_bn_stats(const float* __restrict__ X,
                                                   float* __restrict__ sum,
                                                   float* __restrict__ sq, int M) {
  __shared__ float ls[128], lq[128];
  const int tid = threadIdx.x;
  const int c = tid & 127, rh = tid >> 7;
  const int rb = blockIdx.x * 128;
  float p1 = 0.f, p2 = 0.f;
  for (int r = rh; r < 128; r += 2) {
    int gr = rb + r;
    if (gr < M) {
      float v = X[(size_t)gr * 128 + c];
      p1 += v; p2 += v * v;
    }
  }
  if (tid < 128) { ls[tid] = 0.f; lq[tid] = 0.f; }
  __syncthreads();
  atomicAdd(&ls[c], p1);
  atomicAdd(&lq[c], p2);
  __syncthreads();
  if (tid < 128) {
    atomicAdd(&sum[tid], ls[tid]);
    atomicAdd(&sq[tid], lq[tid]);
  }
}

// hn = silu(bn(out)) + sc
__global__ void mf_bn_final(const float* __restrict__ X, const float* __restrict__ sc,
                            const float* __restrict__ sum, const float* __restrict__ sq,
                            const float* __restrict__ g, const float* __restrict__ b,
                            float* __restrict__ hn, int M) {
  int i = blockIdx.x * blockDim.x + threadIdx.x;
  if (i >= M * 128) return;
  int c = i & 127;
  float inv = 1.f / (float)M;
  float mean = sum[c] * inv;
  float var  = sq[c] * inv - mean * mean;
  float v = (X[i] - mean) * rsqrtf(var + 1e-5f) * g[c] + b[c];
  hn[i] = v * sigmoidf_(v) + sc[i];
}

__global__ void mf_pool(const float* __restrict__ h, const int* __restrict__ batch,
                        float* __restrict__ pooled, float* __restrict__ counts) {
  int n = blockIdx.x * blockDim.x + threadIdx.x;
  if (n >= NN) return;
  int bg = batch[n];
  atomicAdd(&counts[bg], 1.f);
  const float* row = h + (size_t)n * FDIM;
  float* prow = pooled + (size_t)bg * FDIM;
  for (int c = 0; c < FDIM; ++c) atomicAdd(&prow[c], row[c]);
}

__global__ void mf_pool_div(float* __restrict__ pooled, const float* __restrict__ counts) {
  int g = blockIdx.x, c = threadIdx.x;
  pooled[(size_t)g * FDIM + c] /= fmaxf(counts[g], 1.f);
}

// ---------------------------------------------------------------- host orchestration
extern "C" void kernel_launch(void* const* d_in, const int* in_sizes, int n_in,
                              void* d_out, int out_size, void* d_ws, size_t ws_size,
                              hipStream_t stream) {
  (void)in_sizes; (void)n_in; (void)out_size; (void)ws_size;

  const float* x         = (const float*)d_in[0];
  const float* edge_attr = (const float*)d_in[1];
  const int*   eidx      = (const int*)d_in[2];
  const int*   batch     = (const int*)d_in[3];
  const int*   task_id   = (const int*)d_in[4];
  const int*   src = eidx;
  const int*   dst = eidx + NE;

  const float* P_embW = (const float*)d_in[5];
  const float* P_embb = (const float*)d_in[6];
  const float* P_rW1  = (const float*)d_in[7];
  const float* P_rb1  = (const float*)d_in[8];
  const float* P_rW2  = (const float*)d_in[9];
  const float* P_rb2  = (const float*)d_in[10];
  const float* P_Wq   = (const float*)d_in[11];
  const float* P_bq   = (const float*)d_in[12];
  const float* P_Wk   = (const float*)d_in[13];
  const float* P_bk   = (const float*)d_in[14];
  const float* P_Wv   = (const float*)d_in[15];
  const float* P_bv   = (const float*)d_in[16];
  const float* P_We   = (const float*)d_in[17];
  const float* P_Wmu  = (const float*)d_in[18];
  const float* P_bmu  = (const float*)d_in[19];
  const float* P_Wml  = (const float*)d_in[20];
  const float* P_bml  = (const float*)d_in[21];
  const float* P_lag  = (const float*)d_in[22];
  const float* P_lab  = (const float*)d_in[23];
  const float* P_lmg  = (const float*)d_in[24];
  const float* P_lmb  = (const float*)d_in[25];
  const float* P_Wc   = (const float*)d_in[26];
  const float* P_bc   = (const float*)d_in[27];
  const float* P_bng  = (const float*)d_in[28];
  const float* P_bnb  = (const float*)d_in[29];
  const float* P_Ws   = (const float*)d_in[30];
  const float* P_bs   = (const float*)d_in[31];
  const float* P_temb = (const float*)d_in[32];
  const float* P_fcW  = (const float*)d_in[33];
  const float* P_fcb  = (const float*)d_in[34];

  // workspace bump allocator
  char* wp = (char*)d_ws;
  auto alloc = [&](size_t bytes) -> char* {
    char* p = wp;
    wp += (bytes + 255) & ~(size_t)255;
    return p;
  };
  float* hA     = (float*)alloc((size_t)NN * FDIM * 4);
  float* hB     = (float*)alloc((size_t)NN * FDIM * 4);
  float* ebuf   = (float*)alloc((size_t)NE * FDIM * 4);
  float* outb   = (float*)alloc((size_t)NE * FDIM * 4);   // rbf tmp + per-layer pre-BN out
  u16*   qbuf   = (u16*)alloc((size_t)NN * HC * 2);
  u16*   kbuf   = (u16*)alloc((size_t)NN * HC * 2);
  u16*   vbuf   = (u16*)alloc((size_t)NN * HC * 2);
  u16*   eabuf  = (u16*)alloc((size_t)NE * HC * 2);
  float* agg    = (float*)alloc((size_t)NN * HC * 4);
  float* scb    = (float*)alloc((size_t)NN * FDIM * 4);
  float* bnsum  = (float*)alloc(512);
  float* bnsq   = (float*)alloc(512);
  float* pooled = (float*)alloc((size_t)NG * FDIM * 4);
  float* counts = (float*)alloc((size_t)NG * 4);
  // pre-transposed bf16 weights
  u16* embWT = (u16*)alloc((size_t)128 * 96 * 2);
  u16* rW1T  = (u16*)alloc((size_t)128 * 128 * 2);
  u16* rW2T  = (u16*)alloc((size_t)128 * 128 * 2);
  u16* WqT   = (u16*)alloc((size_t)NLAYERS * HC * FDIM * 2);
  u16* WkT   = (u16*)alloc((size_t)NLAYERS * HC * FDIM * 2);
  u16* WvT   = (u16*)alloc((size_t)NLAYERS * HC * FDIM * 2);
  u16* WeT   = (u16*)alloc((size_t)NLAYERS * HC * FDIM * 2);
  u16* WmuT  = (u16*)alloc((size_t)NLAYERS * TC * TC * 2);
  u16* WmlT  = (u16*)alloc((size_t)NLAYERS * CC * TC * 2);
  u16* WcT   = (u16*)alloc((size_t)NLAYERS * FDIM * HC * 2);
  u16* WsT   = (u16*)alloc((size_t)NLAYERS * FDIM * FDIM * 2);
  u16* fcWT  = (u16*)alloc((size_t)128 * 128 * 2);

  auto TR = [&](const float* W, u16* Wt, int K, int N, int Kpad) {
    mf_tr<<<(N * Kpad + 255) / 256, 256, 0, stream>>>(W, Wt, K, N, Kpad);
  };
  TR(P_embW, embWT, ATOMF, FDIM, 96);
  TR(P_rW1, rW1T, FDIM, FDIM, FDIM);
  TR(P_rW2, rW2T, FDIM, FDIM, FDIM);
  TR(P_fcW, fcWT, FDIM, FDIM, FDIM);
  for (int i = 0; i < NLAYERS; ++i) {
    TR(P_Wq + (size_t)i * FDIM * HC, WqT + (size_t)i * HC * FDIM, FDIM, HC, FDIM);
    TR(P_Wk + (size_t)i * FDIM * HC, WkT + (size_t)i * HC * FDIM, FDIM, HC, FDIM);
    TR(P_Wv + (size_t)i * FDIM * HC, WvT + (size_t)i * HC * FDIM, FDIM, HC, FDIM);
    TR(P_We + (size_t)i * FDIM * HC, WeT + (size_t)i * HC * FDIM, FDIM, HC, FDIM);
    TR(P_Wmu + (size_t)i * TC * TC, WmuT + (size_t)i * TC * TC, TC, TC, TC);
    TR(P_Wml + (size_t)i * TC * CC, WmlT + (size_t)i * CC * TC, TC, CC, TC);
    TR(P_Wc + (size_t)i * HC * FDIM, WcT + (size_t)i * FDIM * HC, HC, FDIM, HC);
    TR(P_Ws + (size_t)i * FDIM * FDIM, WsT + (size_t)i * FDIM * FDIM, FDIM, FDIM, FDIM);
  }

  auto gemm_grid = [](int M, int N) { return dim3((unsigned)(N / 128), (unsigned)((M + 31) / 32)); };

  // h = x @ emb_W + emb_b            (K=92, padded to 96)
  mf_gemm<0, false><<<gemm_grid(NN, FDIM), 256, 0, stream>>>(x, embWT, P_embb, hA, NN, FDIM, ATOMF, 96);
  // e = softplus(ea @ W1 + b1) @ W2 + b2
  mf_gemm<1, false><<<gemm_grid(NE, FDIM), 256, 0, stream>>>(edge_attr, rW1T, P_rb1, outb, NE, FDIM, FDIM, FDIM);
  mf_gemm<0, false><<<gemm_grid(NE, FDIM), 256, 0, stream>>>(outb, rW2T, P_rb2, ebuf, NE, FDIM, FDIM, FDIM);

  float* hcur = hA;
  float* hnxt = hB;

  for (int i = 0; i < NLAYERS; ++i) {
    if (i == 2)
      mf_task_add<<<(NN * FDIM + 255) / 256, 256, 0, stream>>>(hcur, P_temb, task_id);

    const float* bq = P_bq + (size_t)i * HC;
    const float* bk = P_bk + (size_t)i * HC;
    const float* bv = P_bv + (size_t)i * HC;
    const float* bmu = P_bmu + (size_t)i * TC;
    const float* bml = P_bml + (size_t)i * CC;
    const float* lag = P_lag + (size_t)i * TC;  const float* lab = P_lab + (size_t)i * TC;
    const float* lmg = P_lmg + (size_t)i * CC;  const float* lmb = P_lmb + (size_t)i * CC;
    const float* bc = P_bc + (size_t)i * FDIM;
    const float* bng = P_bng + (size_t)i * FDIM; const float* bnb = P_bnb + (size_t)i * FDIM;
    const float* bs = P_bs + (size_t)i * FDIM;

    // projections (bf16 outputs feed WMMA + gathers)
    mf_gemm<0, true><<<gemm_grid(NN, HC), 256, 0, stream>>>(hcur, WqT + (size_t)i * HC * FDIM, bq, qbuf, NN, HC, FDIM, FDIM);
    mf_gemm<0, true><<<gemm_grid(NN, HC), 256, 0, stream>>>(hcur, WkT + (size_t)i * HC * FDIM, bk, kbuf, NN, HC, FDIM, FDIM);
    mf_gemm<0, true><<<gemm_grid(NN, HC), 256, 0, stream>>>(hcur, WvT + (size_t)i * HC * FDIM, bv, vbuf, NN, HC, FDIM, FDIM);
    mf_gemm<0, true><<<gemm_grid(NE, HC), 256, 0, stream>>>(ebuf, WeT + (size_t)i * HC * FDIM, nullptr, eabuf, NE, HC, FDIM, FDIM);
    // residual skip: sc = h @ Ws + bs
    mf_gemm<0, false><<<gemm_grid(NN, FDIM), 256, 0, stream>>>(hcur, WsT + (size_t)i * FDIM * FDIM, bs, scb, NN, FDIM, FDIM, FDIM);

    mf_zero<<<2048, 256, 0, stream>>>(agg, NN * HC);
    mf_edge<<<NE / 8, 256, 0, stream>>>(src, dst, qbuf, kbuf, vbuf, eabuf,
                                        WmuT + (size_t)i * TC * TC, bmu,
                                        WmlT + (size_t)i * CC * TC, bml,
                                        lag, lab, lmg, lmb, agg);

    // out = agg @ Wc + bc; hn = silu(bn(out)) + sc
    mf_gemm<0, false><<<gemm_grid(NN, FDIM), 256, 0, stream>>>(agg, WcT + (size_t)i * FDIM * HC, bc, outb, NN, FDIM, HC, HC);
    mf_zero<<<1, 256, 0, stream>>>(bnsum, 128);
    mf_zero<<<1, 256, 0, stream>>>(bnsq, 128);
    mf_bn_stats<<<(NN + 127) / 128, 256, 0, stream>>>(outb, bnsum, bnsq, NN);
    mf_bn_final<<<(NN * FDIM + 255) / 256, 256, 0, stream>>>(outb, scb, bnsum, bnsq,
                                                             bng, bnb, hnxt, NN);
    float* t = hcur; hcur = hnxt; hnxt = t;
  }

  // graph mean-pool + final FC + silu
  mf_zero<<<64, 256, 0, stream>>>(pooled, NG * FDIM);
  mf_zero<<<1, 256, 0, stream>>>(counts, NG);
  mf_pool<<<(NN + 255) / 256, 256, 0, stream>>>(hcur, batch, pooled, counts);
  mf_pool_div<<<NG, FDIM, 0, stream>>>(pooled, counts);
  mf_gemm<2, false><<<gemm_grid(NG, FDIM), 256, 0, stream>>>(pooled, fcWT, P_fcb,
                                                             (float*)d_out, NG, FDIM, FDIM, FDIM);
}
